// Block_88983132438844
// MI455X (gfx1250) — matrix operand order repaired and checked by reading, not compiled, not run on hardware
//
#include <hip/hip_runtime.h>
#include <stdint.h>

// RWKV-4 block for MI455X (gfx1250, wave32).
// GEMMs: v_wmma_f32_16x16x32_bf16 (bf16 A/B, f32 accumulate), LDS double-buffered
// with GLOBAL_LOAD_ASYNC_TO_LDS_B128 (ASYNCcnt-tracked memory->LDS DMA) so HBM
// traffic overlaps the matrix pipe. ~0.89 TFLOP GEMM work vs ~0.5GB traffic.

typedef __bf16 bf16;
typedef __bf16 v8bf  __attribute__((ext_vector_type(8)));
typedef __bf16 v16bf __attribute__((ext_vector_type(16)));
typedef float  v8f   __attribute__((ext_vector_type(8)));

#define B_  4
#define T_  2048
#define C_  2048
#define D_  2048
#define F_  8192
#define BT_ 8192
#define EPS_ 1e-5f

// ---------------- f32 -> bf16 weight conversion ----------------
__global__ __launch_bounds__(256) void cvt_bf16_kernel(const float* __restrict__ in,
                                                       bf16* __restrict__ out, int n)
{
    int i = blockIdx.x * 256 + threadIdx.x;
    int stride = gridDim.x * 256;
    for (; i < n; i += stride) out[i] = (bf16)in[i];
}

// ---------------- bf16 WMMA GEMM:  out[M,N] = A[M,K] * W[N,K]^T ----------------
// 256 threads = 8 waves (4x2), block tile 128x128, BK=64, double-buffered LDS
// filled by async global->LDS b128 copies.
// EP epilogues: 0 = store f32
//               1 = store f32 (aux1 + acc)                      [residual add]
//               2 = store bf16 relu(acc)^2                      [ChannelMix k]
//               3 = store f32 (aux1 + sigmoid(aux2) * acc)      [final output]
#define BM 128
#define BN 128
#define BK 64
#define LDT 80   // padded LDS row length in halfs (160B: keeps v16bf reads 32B-aligned)

union AFrag { v16bf v; v8bf h[2]; };

template<int EP>
__global__ __launch_bounds__(256) void gemm_bf16_kernel(
    const bf16* __restrict__ A, const bf16* __restrict__ W,
    int N, int K,
    float* __restrict__ outF, bf16* __restrict__ outH,
    const float* __restrict__ aux1, const float* __restrict__ aux2)
{
    __shared__ alignas(32) bf16 As[2][BM * LDT];
    __shared__ alignas(32) bf16 Bs[2][BN * LDT];

    const int tid  = threadIdx.x;
    const int lane = tid & 31;
    const int wv   = tid >> 5;   // wave 0..7
    const int wm   = wv >> 1;    // 0..3 : 32 rows each
    const int wn   = wv & 1;     // 0..1 : 64 cols each
    const int lm   = lane & 15;
    const int hi   = lane >> 4;
    const size_t gm = (size_t)blockIdx.y * BM;
    const size_t gn = (size_t)blockIdx.x * BN;

    v8f acc[2][4];
    #pragma unroll
    for (int i = 0; i < 2; ++i)
        #pragma unroll
        for (int j = 0; j < 4; ++j)
            #pragma unroll
            for (int e = 0; e < 8; ++e) acc[i][j][e] = 0.f;

    // issue 8 async global->LDS b128 copies for one 128x64-half A tile + W tile
    auto issue = [&](int k0, int buf) {
        #pragma unroll
        for (int i = 0; i < 4; ++i) {
            int id = tid + i * 256;
            int r  = id >> 3;
            int c  = (id & 7) * 8;
            uint64_t ga = (uint64_t)(uintptr_t)(A + (gm + r) * (size_t)K + k0 + c);
            uint64_t gb = (uint64_t)(uintptr_t)(W + (gn + r) * (size_t)K + k0 + c);
            uint32_t la = (uint32_t)(uintptr_t)(&As[buf][r * LDT + c]);
            uint32_t lb = (uint32_t)(uintptr_t)(&Bs[buf][r * LDT + c]);
            asm volatile("global_load_async_to_lds_b128 %0, %1, off"
                         :: "v"(la), "v"(ga) : "memory");
            asm volatile("global_load_async_to_lds_b128 %0, %1, off"
                         :: "v"(lb), "v"(gb) : "memory");
        }
    };

    issue(0, 0);
    const int nt = K / BK;
    for (int it = 0; it < nt; ++it) {
        const int buf = it & 1;
        if (it + 1 < nt) {
            issue((it + 1) * BK, buf ^ 1);
            // 8 ops in flight belong to tile it+1; async loads retire in order,
            // so ASYNCcnt <= 8  =>  tile `it` fully landed in LDS.
            asm volatile("s_wait_asynccnt 0x8" ::: "memory");
        } else {
            asm volatile("s_wait_asynccnt 0x0" ::: "memory");
        }
        __syncthreads();

        #pragma unroll
        for (int ks = 0; ks < BK / 32; ++ks) {
            // A fragment: lane holds row m; elements = 16 consecutive K (per lane half)
            AFrag a[2];
            #pragma unroll
            for (int tm = 0; tm < 2; ++tm) {
                const bf16* p = &As[buf][(wm * 32 + tm * 16 + lm) * LDT + ks * 32 + hi * 16];
                a[tm].h[0] = *(const v8bf*)p;            // K = hi*16 .. hi*16+7
                a[tm].h[1] = *(const v8bf*)(p + 8);      // K = hi*16+8 .. hi*16+15
            }
            // B fragment: lane holds column n; elements = 16 consecutive K (per lane half)
            v16bf b[4];
            #pragma unroll
            for (int tn = 0; tn < 4; ++tn)
                b[tn] = *(const v16bf*)(&Bs[buf][(wn * 64 + tn * 16 + lm) * LDT + ks * 32 + hi * 16]);

            #pragma unroll
            for (int tm = 0; tm < 2; ++tm)
                #pragma unroll
                for (int tn = 0; tn < 4; ++tn)
                    acc[tm][tn] = __builtin_amdgcn_wmma_f32_16x16x32_bf16(
                        false, a[tm].v, false, b[tn], (short)0, acc[tm][tn], false, false);
        }
        __syncthreads();   // all waves done with As[buf]/Bs[buf] before it's refilled
    }

    // epilogue: C/D layout = VGPR j -> row (j + 8*hi), col = lane%16
    #pragma unroll
    for (int tm = 0; tm < 2; ++tm) {
        size_t rbase = gm + wm * 32 + tm * 16 + hi * 8;
        #pragma unroll
        for (int tn = 0; tn < 4; ++tn) {
            size_t col = gn + wn * 64 + tn * 16 + lm;
            #pragma unroll
            for (int j = 0; j < 8; ++j) {
                size_t idx = (rbase + j) * (size_t)N + col;
                float v = acc[tm][tn][j];
                if (EP == 0) {
                    outF[idx] = v;
                } else if (EP == 1) {
                    outF[idx] = aux1[idx] + v;
                } else if (EP == 2) {
                    float t = v > 0.f ? v : 0.f;
                    outH[idx] = (bf16)(t * t);
                } else {
                    float s = 1.f / (1.f + __expf(-aux2[idx]));
                    outF[idx] = aux1[idx] + s * v;
                }
            }
        }
    }
}

// ---------------- fused LayerNorm + time_shift + lerp-mix -> bf16 ----------------
// One block per (b,t) row; recomputes LN of row t-1 (cheap, avoids staging pass).
template<bool THREE>
__global__ __launch_bounds__(256) void ln_mix_kernel(
    const float* __restrict__ x,
    const float* __restrict__ lw, const float* __restrict__ lb,
    const float* __restrict__ mk, const float* __restrict__ mv, const float* __restrict__ mr,
    bf16* __restrict__ ok, bf16* __restrict__ ov, bf16* __restrict__ orr)
{
    const int bt  = blockIdx.x;
    const int t   = bt & (T_ - 1);
    const int tid = threadIdx.x;
    const size_t rowoff = (size_t)bt * C_ + tid * 8;
    const float* xt = x + rowoff;
    const float* xp = xt - C_;

    float vt[8], vp[8];
    {
        float4 a0 = ((const float4*)xt)[0];
        float4 a1 = ((const float4*)xt)[1];
        vt[0]=a0.x; vt[1]=a0.y; vt[2]=a0.z; vt[3]=a0.w;
        vt[4]=a1.x; vt[5]=a1.y; vt[6]=a1.z; vt[7]=a1.w;
    }
    if (t > 0) {
        float4 a0 = ((const float4*)xp)[0];
        float4 a1 = ((const float4*)xp)[1];
        vp[0]=a0.x; vp[1]=a0.y; vp[2]=a0.z; vp[3]=a0.w;
        vp[4]=a1.x; vp[5]=a1.y; vp[6]=a1.z; vp[7]=a1.w;
    } else {
        #pragma unroll
        for (int j = 0; j < 8; ++j) vp[j] = 0.f;
    }

    float s_t = 0.f, q_t = 0.f, s_p = 0.f, q_p = 0.f;
    #pragma unroll
    for (int j = 0; j < 8; ++j) {
        s_t += vt[j]; q_t += vt[j] * vt[j];
        s_p += vp[j]; q_p += vp[j] * vp[j];
    }

    __shared__ float4 red[256];
    red[tid] = make_float4(s_t, q_t, s_p, q_p);
    __syncthreads();
    for (int off = 128; off > 0; off >>= 1) {
        if (tid < off) {
            float4 o = red[tid + off];
            red[tid].x += o.x; red[tid].y += o.y;
            red[tid].z += o.z; red[tid].w += o.w;
        }
        __syncthreads();
    }
    float4 tot = red[0];

    const float invC = 1.f / (float)C_;
    float mu_t = tot.x * invC;
    float rs_t = rsqrtf(tot.y * invC - mu_t * mu_t + EPS_);
    float mu_p = tot.z * invC;
    float rs_p = rsqrtf(tot.w * invC - mu_p * mu_p + EPS_);

    #pragma unroll
    for (int j = 0; j < 8; ++j) {
        int c = tid * 8 + j;
        float w = lw[c], b = lb[c];
        float h  = (vt[j] - mu_t) * rs_t * w + b;
        float hh = (t > 0) ? (vp[j] - mu_p) * rs_p * w + b : 0.f;
        float ak = mk[c];
        ok[rowoff + j] = (bf16)(h * ak + hh * (1.f - ak));
        if (THREE) {
            float av = mv[c];
            ov[rowoff + j] = (bf16)(h * av + hh * (1.f - av));
        }
        float ar = mr[c];
        orr[rowoff + j] = (bf16)(h * ar + hh * (1.f - ar));
    }
}

// ---------------- WKV linear recurrence (numerically-stable), fused sigmoid(r)* ----------------
__global__ __launch_bounds__(256) void wkv_kernel(
    const float* __restrict__ kk, const float* __restrict__ vv, const float* __restrict__ rr,
    const float* __restrict__ time_decay, const float* __restrict__ time_first,
    bf16* __restrict__ out)
{
    int gi = blockIdx.x * 256 + threadIdx.x;  // 0 .. B*D-1
    int b  = gi / D_;
    int d  = gi - b * D_;
    float decay = -__expf(time_decay[d]);
    float tf    = time_first[d];
    float aa = 0.f, bb = 0.f, pp = -1e38f;
    size_t base = (size_t)b * T_ * D_ + d;
    for (int t = 0; t < T_; ++t) {
        size_t idx = base + (size_t)t * D_;
        float kt = kk[idx], vt = vv[idx];
        float ww = tf + kt;
        float p  = fmaxf(pp, ww);
        float e1 = __expf(pp - p);
        float e2 = __expf(ww - p);
        float o  = (e1 * aa + e2 * vt) / (e1 * bb + e2);
        float ww2 = pp + decay;
        float p2  = fmaxf(ww2, kt);
        float e1b = __expf(ww2 - p2);
        float e2b = __expf(kt - p2);
        aa = e1b * aa + e2b * vt;
        bb = e1b * bb + e2b;
        pp = p2;
        float sr = 1.f / (1.f + __expf(-rr[idx]));
        out[idx] = (bf16)(sr * o);
    }
}

// ---------------- orchestration ----------------
extern "C" void kernel_launch(void* const* d_in, const int* in_sizes, int n_in,
                              void* d_out, int out_size, void* d_ws, size_t ws_size,
                              hipStream_t stream)
{
    const float* x          = (const float*)d_in[0];
    const float* ln1_w      = (const float*)d_in[1];
    const float* ln1_b      = (const float*)d_in[2];
    const float* ln2_w      = (const float*)d_in[3];
    const float* ln2_b      = (const float*)d_in[4];
    const float* tm_k       = (const float*)d_in[5];
    const float* tm_v       = (const float*)d_in[6];
    const float* tm_r       = (const float*)d_in[7];
    const float* time_decay = (const float*)d_in[8];
    const float* time_first = (const float*)d_in[9];
    const float* Wk         = (const float*)d_in[10];
    const float* Wv         = (const float*)d_in[11];
    const float* Wr         = (const float*)d_in[12];
    const float* Wo         = (const float*)d_in[13];
    const float* cm_k       = (const float*)d_in[14];
    const float* cm_r       = (const float*)d_in[15];
    const float* Wck        = (const float*)d_in[16];
    const float* Wcv        = (const float*)d_in[17];
    const float* Wcr        = (const float*)d_in[18];

    char* ws = (char*)d_ws;
    const size_t MB = 1ull << 20;
    // bf16 weights: 104MB
    bf16* Wk_h  = (bf16*)(ws + 0 * MB);
    bf16* Wv_h  = (bf16*)(ws + 8 * MB);
    bf16* Wr_h  = (bf16*)(ws + 16 * MB);
    bf16* Wo_h  = (bf16*)(ws + 24 * MB);
    bf16* Wcr_h = (bf16*)(ws + 32 * MB);
    bf16* Wck_h = (bf16*)(ws + 40 * MB);
    bf16* Wcv_h = (bf16*)(ws + 72 * MB);
    // bf16 mixed activations: 96MB (reused later)
    bf16* xk_h  = (bf16*)(ws + 104 * MB);
    bf16* xv_h  = (bf16*)(ws + 136 * MB);
    bf16* xr_h  = (bf16*)(ws + 168 * MB);
    // f32 k/v/r: 192MB (reused later)
    float* k_f  = (float*)(ws + 200 * MB);
    float* v_f  = (float*)(ws + 264 * MB);
    float* r_f  = (float*)(ws + 328 * MB);
    // f32 post-TimeMix residual: 64MB   (peak ws = 456MB)
    float* x1_f = (float*)(ws + 392 * MB);
    // overlays (lifetimes disjoint):
    bf16* rwkv_h = xr_h;                 // written after xr consumed
    bf16* xk2_h  = xk_h;
    bf16* xr2_h  = xv_h;
    bf16* kf_h   = (bf16*)(ws + 200 * MB);  // 128MB over k_f+v_f (consumed by wkv)
    float* r2_f  = r_f;                     // r consumed by wkv

    auto cvt = [&](const float* src, bf16* dst, int n) {
        int blocks = (n + 255) / 256;
        if (blocks > 16384) blocks = 16384;
        cvt_bf16_kernel<<<blocks, 256, 0, stream>>>(src, dst, n);
    };
    cvt(Wk,  Wk_h,  D_ * C_);
    cvt(Wv,  Wv_h,  D_ * C_);
    cvt(Wr,  Wr_h,  D_ * C_);
    cvt(Wo,  Wo_h,  C_ * D_);
    cvt(Wcr, Wcr_h, C_ * C_);
    cvt(Wck, Wck_h, F_ * C_);
    cvt(Wcv, Wcv_h, C_ * F_);

    dim3 g16(C_ / BN, BT_ / BM);   // N=2048 GEMMs
    dim3 g64(F_ / BN, BT_ / BM);   // N=8192 GEMM

    // --- TimeMix ---
    ln_mix_kernel<true><<<BT_, 256, 0, stream>>>(x, ln1_w, ln1_b, tm_k, tm_v, tm_r,
                                                 xk_h, xv_h, xr_h);
    gemm_bf16_kernel<0><<<g16, 256, 0, stream>>>(xk_h, Wk_h, D_, C_, k_f, nullptr, nullptr, nullptr);
    gemm_bf16_kernel<0><<<g16, 256, 0, stream>>>(xv_h, Wv_h, D_, C_, v_f, nullptr, nullptr, nullptr);
    gemm_bf16_kernel<0><<<g16, 256, 0, stream>>>(xr_h, Wr_h, D_, C_, r_f, nullptr, nullptr, nullptr);
    wkv_kernel<<<(B_ * D_) / 256, 256, 0, stream>>>(k_f, v_f, r_f, time_decay, time_first, rwkv_h);
    gemm_bf16_kernel<1><<<g16, 256, 0, stream>>>(rwkv_h, Wo_h, C_, D_, x1_f, nullptr, x, nullptr);

    // --- ChannelMix ---
    ln_mix_kernel<false><<<BT_, 256, 0, stream>>>(x1_f, ln2_w, ln2_b, cm_k, nullptr, cm_r,
                                                  xk2_h, nullptr, xr2_h);
    gemm_bf16_kernel<2><<<g64, 256, 0, stream>>>(xk2_h, Wck_h, F_, C_, nullptr, kf_h, nullptr, nullptr);
    gemm_bf16_kernel<0><<<g16, 256, 0, stream>>>(xr2_h, Wcr_h, C_, C_, r2_f, nullptr, nullptr, nullptr);
    gemm_bf16_kernel<3><<<g16, 256, 0, stream>>>(kf_h, Wcv_h, C_, F_, (float*)d_out, nullptr,
                                                 x1_f, r2_f);
}
